// Bidirectional_GRU_Encoder_21234318311659
// MI455X (gfx1250) — matrix operand order repaired and checked by reading, not compiled, not run on hardware
//
#include <hip/hip_runtime.h>

// Bidirectional 2-layer GRU: B=64, T=1024, I=H=512.
//  - Weights/x converted to bf16 once.
//  - gi = x@WihT + bih precomputed with a massively parallel WMMA GEMM.
//  - Recurrence: 1 persistent workgroup per (layer,dir); h in LDS
//    (fp32 master 128KB + bf16 copy 64KB); per step 64x1536x512 bf16 WMMA GEMM.
//  - Gate-triple mapping keeps r/z/n for the same columns in matching lanes.
//  - v3: software-pipelined global B fragments (prefetch k+1 during k's WMMAs).
//  - v5: async global->LDS staging via __builtin_amdgcn_global_load_async_to_lds_b128
//    (probe showed param0 is v4i* generic; LDS side is address_space(3)).

#define TT 1024
#define BB 64
#define II 512
#define HH 512
#define G3 1536
#define LL 2

#if defined(__has_builtin)
#if __has_builtin(__builtin_amdgcn_global_load_async_to_lds_b128)
#define HAVE_ASYNC_LDS 1
#endif
#endif
#ifndef HAVE_ASYNC_LDS
#define HAVE_ASYNC_LDS 0
#endif

typedef __attribute__((ext_vector_type(16))) __bf16 v16bf;
typedef __attribute__((ext_vector_type(8)))  float  v8f;
typedef __attribute__((ext_vector_type(4)))  int    v4i;
typedef __attribute__((address_space(3)))    v4i    lds_v4i;

union BF16x16 { uint4 q[2]; v16bf v; };

__device__ inline unsigned short f2bf(float f) {
  unsigned int x = __float_as_uint(f);
  unsigned int r = x + 0x7FFFu + ((x >> 16) & 1u);  // round-to-nearest-even
  return (unsigned short)(r >> 16);
}

// B fragment: 16 contiguous bf16 (32B) per lane.
__device__ inline v16bf load_fragB(const unsigned short* p) {
  BF16x16 u;
  const uint4* q = (const uint4*)p;
  u.q[0] = q[0];
  u.q[1] = q[1];
  return u.v;
}

// A fragment: 16 contiguous bf16 (32B) per lane.
__device__ inline v16bf load_fragA(const unsigned short* p) {
  BF16x16 u;
  const uint4* q = (const uint4*)p;
  u.q[0] = q[0];
  u.q[1] = q[1];
  return u.v;
}

__device__ inline v8f wmma_bf16(v16bf a, v16bf b, v8f c) {
  return __builtin_amdgcn_wmma_f32_16x16x32_bf16(false, a, false, b, (short)0, c,
                                                 false, false);
}

__device__ inline float sigmoidf_(float x) { return 1.0f / (1.0f + __expf(-x)); }
__device__ inline float tanhf_(float x) {
  float e = __expf(-2.0f * x);
  return (1.0f - e) / (1.0f + e);
}

// ---------------------------------------------------------------------------
__global__ void conv_f32_to_bf16(const float* __restrict__ src,
                                 unsigned short* __restrict__ dst, size_t n) {
  size_t i = (size_t)blockIdx.x * blockDim.x + threadIdx.x;
  size_t stride = (size_t)gridDim.x * blockDim.x;
  for (; i < n; i += stride) dst[i] = f2bf(src[i]);
}

// ---------------------------------------------------------------------------
// gi[s][b][0:1536] = inp_row(b, s_eff) @ W^T + bih   (one workgroup per (s,dir))
__global__ __launch_bounds__(1024)
void gi_gemm_kernel(const unsigned short* __restrict__ inp_f,
                    const unsigned short* __restrict__ inp_b,
                    long long strideS, long long strideB, int reverse_b,
                    const unsigned short* __restrict__ W_f,
                    const unsigned short* __restrict__ W_b,
                    const float* __restrict__ bih_f,
                    const float* __restrict__ bih_b,
                    _Float16* __restrict__ gi_f,
                    _Float16* __restrict__ gi_b) {
  extern __shared__ unsigned short sA[];  // 64 x 512 bf16 = 64 KB
  const int s = blockIdx.x, dir = blockIdx.y;
  const unsigned short* inp = dir ? inp_b : inp_f;
  const unsigned short* W   = dir ? W_b   : W_f;
  const float* bih          = dir ? bih_b : bih_f;
  _Float16* gi              = dir ? gi_b  : gi_f;
  const long long s_eff = (dir && reverse_b) ? (long long)(TT - 1 - s) : (long long)s;
  const int tid = threadIdx.x;

  // Stage 64 input rows into LDS (async direct-to-LDS when available).
#pragma unroll
  for (int i = 0; i < 4; ++i) {
    int e = tid + i * 1024;
    int row = e >> 6;
    int c = (e & 63) * 8;
    const unsigned short* src =
        inp + (size_t)row * strideB + (size_t)s_eff * strideS + c;
#if HAVE_ASYNC_LDS
    __builtin_amdgcn_global_load_async_to_lds_b128(
        (v4i*)(unsigned short*)src,
        (lds_v4i*)(sA + row * 512 + c),
        /*offset=*/0, /*cpol=*/0);
#else
    *((uint4*)(sA + row * 512 + c)) = *((const uint4*)src);
#endif
  }
#if HAVE_ASYNC_LDS
#if __has_builtin(__builtin_amdgcn_s_wait_asynccnt)
  __builtin_amdgcn_s_wait_asynccnt(0);
#else
  asm volatile("s_wait_asynccnt 0x0" ::: "memory");
#endif
#endif
  __syncthreads();

  const int w = tid >> 5, lane = tid & 31;
  const int mt = w & 3, nset = w >> 2;     // 4 m-tiles x 8 n-sets (12 tiles each)
  const int m0 = mt * 16;
  const int l15 = lane & 15, lh = lane >> 4;

  const unsigned short* const aBase = sA + (m0 + l15) * 512 + lh * 16;
  const unsigned short* const wBase =
      W + (size_t)(nset * 12 * 16 + l15) * 512 + lh * 16;

  const v8f zero8 = {0.f, 0.f, 0.f, 0.f, 0.f, 0.f, 0.f, 0.f};

  // 3 chunks of 4 tiles: 4 accumulators + double-buffered B fragments.
#pragma unroll
  for (int c = 0; c < 3; ++c) {
    v8f a0 = zero8, a1 = zero8, a2 = zero8, a3 = zero8;
    const unsigned short* wp = wBase + (size_t)(c * 4) * 16 * 512;
    v16bf b0 = load_fragB(wp + 0 * 16 * 512);
    v16bf b1 = load_fragB(wp + 1 * 16 * 512);
    v16bf b2 = load_fragB(wp + 2 * 16 * 512);
    v16bf b3 = load_fragB(wp + 3 * 16 * 512);
#pragma unroll
    for (int k = 0; k < 16; ++k) {
      v16bf af = load_fragA(aBase + k * 32);
      v16bf n0, n1, n2, n3;
      if (k < 15) {  // prefetch next k while current WMMAs execute
        const unsigned short* wn = wp + (k + 1) * 32;
        n0 = load_fragB(wn + 0 * 16 * 512);
        n1 = load_fragB(wn + 1 * 16 * 512);
        n2 = load_fragB(wn + 2 * 16 * 512);
        n3 = load_fragB(wn + 3 * 16 * 512);
      }
      a0 = wmma_bf16(af, b0, a0);
      a1 = wmma_bf16(af, b1, a1);
      a2 = wmma_bf16(af, b2, a2);
      a3 = wmma_bf16(af, b3, a3);
      if (k < 15) { b0 = n0; b1 = n1; b2 = n2; b3 = n3; }
    }
    v8f accs[4] = {a0, a1, a2, a3};
#pragma unroll
    for (int q = 0; q < 4; ++q) {
      const int col = (nset * 12 + c * 4 + q) * 16 + l15;
      const float bv = bih[col];
#pragma unroll
      for (int v = 0; v < 8; ++v) {
        const int m = m0 + v + 8 * lh;
        gi[((size_t)s * BB + m) * G3 + col] = (_Float16)(accs[q][v] + bv);
      }
    }
  }
}

// ---------------------------------------------------------------------------
// One persistent workgroup per direction (gridDim.x == 2). 32 waves.
// Per step: 4 sequential gate-triples, 3 live accs + double-buffered B frags.
__global__ __launch_bounds__(1024)
void gru_recur_kernel(const unsigned short* __restrict__ Whh_fd,
                      const unsigned short* __restrict__ Whh_bd,
                      const float* __restrict__ bhh_fd,
                      const float* __restrict__ bhh_bd,
                      const _Float16* __restrict__ gi_fd,
                      const _Float16* __restrict__ gi_bd,
                      unsigned short* __restrict__ y_fd,
                      unsigned short* __restrict__ y_bd,
                      float* __restrict__ out, float* __restrict__ hout,
                      int layer, int is_last) {
  extern __shared__ unsigned char smem[];
  float* hS = (float*)smem;                                    // 64x512 f32 (128KB)
  unsigned short* hB = (unsigned short*)(smem + BB * HH * 4);  // 64x512 bf16 (64KB)

  const int dir = blockIdx.x;
  const unsigned short* W = dir ? Whh_bd : Whh_fd;
  const float* bhh        = dir ? bhh_bd : bhh_fd;
  const _Float16* gi      = dir ? gi_bd  : gi_fd;
  unsigned short* yout    = dir ? y_bd   : y_fd;

  const int tid = threadIdx.x;
  const int w = tid >> 5, lane = tid & 31;
  const int l15 = lane & 15, lh = lane >> 4;
  const int mt = w & 3, cgb = (w >> 2) * 4;  // 4 column-groups (triples) per wave

  for (int i = tid; i < BB * HH; i += 1024) { hS[i] = 0.0f; hB[i] = 0; }

  float bb[4][3];
#pragma unroll
  for (int p = 0; p < 4; ++p)
#pragma unroll
    for (int g = 0; g < 3; ++g)
      bb[p][g] = bhh[g * 512 + (cgb + p) * 16 + l15];

  __syncthreads();

  const unsigned short* const aBase = hB + (mt * 16 + l15) * 512 + lh * 16;
  const unsigned short* const wBase0 =
      W + (size_t)(cgb * 16 + l15) * 512 + lh * 16;

  const v8f zero8 = {0.f, 0.f, 0.f, 0.f, 0.f, 0.f, 0.f, 0.f};

  for (int s = 0; s < TT; ++s) {
    const _Float16* gis = gi + (size_t)s * BB * G3;
    float hnew[4][8];

#pragma unroll
    for (int p = 0; p < 4; ++p) {
      v8f ar = zero8, az = zero8, an = zero8;
      const unsigned short* wp = wBase0 + (size_t)p * 16 * 512;
      v16bf br = load_fragB(wp + 0 * 512 * 512);
      v16bf bz = load_fragB(wp + 1 * 512 * 512);
      v16bf bn = load_fragB(wp + 2 * 512 * 512);
#pragma unroll
      for (int k = 0; k < 16; ++k) {
        v16bf af = load_fragA(aBase + k * 32);
        v16bf nr, nz, nn2;
        if (k < 15) {  // prefetch next k
          const unsigned short* wn = wp + (k + 1) * 32;
          nr  = load_fragB(wn + 0 * 512 * 512);
          nz  = load_fragB(wn + 1 * 512 * 512);
          nn2 = load_fragB(wn + 2 * 512 * 512);
        }
        ar = wmma_bf16(af, br, ar);
        az = wmma_bf16(af, bz, az);
        an = wmma_bf16(af, bn, an);
        if (k < 15) { br = nr; bz = nz; bn = nn2; }
      }
      // Gate math for this triple, fully in registers.
      const int j = (cgb + p) * 16 + l15;
#pragma unroll
      for (int v = 0; v < 8; ++v) {
        const int m = mt * 16 + v + 8 * lh;
        const float gr = (float)gis[(size_t)m * G3 + j];
        const float gz = (float)gis[(size_t)m * G3 + 512 + j];
        const float gn = (float)gis[(size_t)m * G3 + 1024 + j];
        const float hold = hS[m * 512 + j];
        const float r = sigmoidf_(gr + ar[v] + bb[p][0]);
        const float z = sigmoidf_(gz + az[v] + bb[p][1]);
        const float nn = tanhf_(gn + r * (an[v] + bb[p][2]));
        hnew[p][v] = (1.0f - z) * nn + z * hold;
      }
    }

    __syncthreads();  // all reads of h complete
#pragma unroll
    for (int p = 0; p < 4; ++p) {
      const int j = (cgb + p) * 16 + l15;
#pragma unroll
      for (int v = 0; v < 8; ++v) {
        const int m = mt * 16 + v + 8 * lh;
        const float hn = hnew[p][v];
        hS[m * 512 + j] = hn;
        hB[m * 512 + j] = f2bf(hn);
        if (is_last) {
          const int t = dir ? (TT - 1 - s) : s;
          out[((size_t)m * TT + t) * (2 * HH) + (size_t)dir * HH + j] = hn;
        } else {
          yout[((size_t)s * BB + m) * HH + j] = f2bf(hn);
        }
        if (s == TT - 1) {
          hout[((size_t)layer * BB + m) * (2 * HH) + (size_t)dir * HH + j] = hn;
        }
      }
    }
    __syncthreads();  // writes visible before next step's reads
  }
}

// ---------------------------------------------------------------------------
extern "C" void kernel_launch(void* const* d_in, const int* in_sizes, int n_in,
                              void* d_out, int out_size, void* d_ws, size_t ws_size,
                              hipStream_t stream) {
  (void)in_sizes; (void)n_in; (void)out_size; (void)ws_size;
  const float* x     = (const float*)d_in[0];
  const float* Wih_f = (const float*)d_in[1];
  const float* Whh_f = (const float*)d_in[2];
  const float* bih_f = (const float*)d_in[3];
  const float* bhh_f = (const float*)d_in[4];
  const float* Wih_b = (const float*)d_in[5];
  const float* Whh_b = (const float*)d_in[6];
  const float* bih_b = (const float*)d_in[7];
  const float* bhh_b = (const float*)d_in[8];

  unsigned char* ws = (unsigned char*)d_ws;
  const size_t W_ELEMS  = (size_t)LL * G3 * II;
  const size_t W_LAYER  = (size_t)G3 * II;
  const size_t X_ELEMS  = (size_t)BB * TT * II;
  const size_t GI_ELEMS = (size_t)TT * BB * G3;
  const size_t Y_ELEMS  = (size_t)TT * BB * HH;

  size_t o = 0;
  unsigned short* Wih_f_bf = (unsigned short*)(ws + o); o += W_ELEMS * 2;
  unsigned short* Whh_f_bf = (unsigned short*)(ws + o); o += W_ELEMS * 2;
  unsigned short* Wih_b_bf = (unsigned short*)(ws + o); o += W_ELEMS * 2;
  unsigned short* Whh_b_bf = (unsigned short*)(ws + o); o += W_ELEMS * 2;
  unsigned short* x_bf     = (unsigned short*)(ws + o); o += X_ELEMS * 2;
  _Float16*       gi_f     = (_Float16*)(ws + o);       o += GI_ELEMS * 2;
  _Float16*       gi_b     = (_Float16*)(ws + o);       o += GI_ELEMS * 2;
  unsigned short* y0f      = (unsigned short*)(ws + o); o += Y_ELEMS * 2;
  unsigned short* y0b      = (unsigned short*)(ws + o); o += Y_ELEMS * 2;

  float* outp  = (float*)d_out;
  float* houtp = outp + (size_t)BB * TT * 2 * HH;

  (void)hipFuncSetAttribute((const void*)gru_recur_kernel,
                            hipFuncAttributeMaxDynamicSharedMemorySize, 196608);
  (void)hipFuncSetAttribute((const void*)gi_gemm_kernel,
                            hipFuncAttributeMaxDynamicSharedMemorySize, 65536);

  conv_f32_to_bf16<<<2048, 256, 0, stream>>>(Wih_f, Wih_f_bf, W_ELEMS);
  conv_f32_to_bf16<<<2048, 256, 0, stream>>>(Whh_f, Whh_f_bf, W_ELEMS);
  conv_f32_to_bf16<<<2048, 256, 0, stream>>>(Wih_b, Wih_b_bf, W_ELEMS);
  conv_f32_to_bf16<<<2048, 256, 0, stream>>>(Whh_b, Whh_b_bf, W_ELEMS);
  conv_f32_to_bf16<<<2048, 256, 0, stream>>>(x, x_bf, X_ELEMS);

  // ---- layer 0 ----
  gi_gemm_kernel<<<dim3(TT, 2), 1024, 65536, stream>>>(
      x_bf, x_bf, (long long)II, (long long)TT * II, /*reverse_b=*/1,
      Wih_f_bf, Wih_b_bf, bih_f, bih_b, gi_f, gi_b);
  gru_recur_kernel<<<2, 1024, 196608, stream>>>(
      Whh_f_bf, Whh_b_bf, bhh_f, bhh_b, gi_f, gi_b, y0f, y0b,
      outp, houtp, /*layer=*/0, /*is_last=*/0);

  // ---- layer 1 ----
  gi_gemm_kernel<<<dim3(TT, 2), 1024, 65536, stream>>>(
      y0f, y0b, (long long)BB * HH, (long long)HH, /*reverse_b=*/0,
      Wih_f_bf + W_LAYER, Wih_b_bf + W_LAYER, bih_f + G3, bih_b + G3,
      gi_f, gi_b);
  gru_recur_kernel<<<2, 1024, 196608, stream>>>(
      Whh_f_bf + W_LAYER, Whh_b_bf + W_LAYER, bhh_f + G3, bhh_b + G3,
      gi_f, gi_b, y0f, y0b, outp, houtp, /*layer=*/1, /*is_last=*/1);
}
